// MLPMoE_53231824666983
// MI455X (gfx1250) — matrix operand order repaired and checked by reading, not compiled, not run on hardware
//
#include <hip/hip_runtime.h>
#include <hip/hip_bf16.h>
#include <math.h>

// ---------------------------------------------------------------------------
// MoE MLP for MI455X (gfx1250).  fp32 reference math reproduced with
// bf16 hi/lo split GEMMs on V_WMMA_F32_16X16X32_BF16 (f32 accumulate).
// MT=32 tokens/block: every B fragment is reused across 2 M-tiles, halving
// L2 weight-fragment traffic vs MT=16.  512-thread blocks (16 waves).
// ---------------------------------------------------------------------------

typedef __attribute__((ext_vector_type(16))) __bf16 v16bf;
typedef __attribute__((ext_vector_type(8)))  float  v8f;

constexpr int B_ = 16, N_ = 2048, D_ = 1024, H_ = 1024, E_ = 8;
constexpr int T_ = B_ * N_;                    // 32768 tokens
constexpr int MT = 32;                         // tokens per MoE tile
constexpr int MAXE = T_ * 2 + E_ * MT;         // padded gathered entries cap (65792)
constexpr int MAXTILES = MAXE / MT;            // 2056
constexpr long long OUTN = (long long)T_ * H_; // result elements

// workspace layout (bytes)
constexpr size_t WS_CNT    = 0;                       // 8  int
constexpr size_t WS_FILL   = 32;                      // 8  int
constexpr size_t WS_OFF    = 64;                      // 9  int
constexpr size_t WS_ZSUM   = 128;                     // 1  float
constexpr size_t WS_DPROX  = 192;                     // 128 float
constexpr size_t WS_DENS1  = 704;                     // 128 float
constexpr size_t WS_TOPKID = 4096;                    // 65536 int
constexpr size_t WS_TOPKW  = WS_TOPKID + 262144;      // 65536 float
constexpr size_t WS_GTOK   = WS_TOPKW  + 262144;      // MAXE int
constexpr size_t WS_GCW    = WS_GTOK   + 264192;      // MAXE float
constexpr size_t WS_WHI    = 2u * 1024u * 1024u;      // packed bf16 hi
constexpr size_t WPACK_ELEMS = (size_t)2 * E_ * 64 * 32 * 32 * 16; // 16.78M
constexpr size_t WS_WLO    = WS_WHI + WPACK_ELEMS * 2;
// total ws requirement ~ WS_WLO + WPACK_ELEMS*2  (~66 MB)

static __device__ __forceinline__ unsigned short bf16_rn(float f) {
  unsigned u = __float_as_uint(f);
  u += 0x7fffu + ((u >> 16) & 1u);
  return (unsigned short)(u >> 16);
}
static __device__ __forceinline__ float bf16_f32(unsigned short h) {
  return __uint_as_float(((unsigned)h) << 16);
}

// ---------------------------------------------------------------------------
// Pack w1/w2 (fp32, [e][k][n]) into per-lane WMMA B-fragment layout, bf16
// hi/lo.  B 32x16 order used by this code (same as A packing): lane L holds column n=L%16, K = (L>=16)*16 + i.
// ---------------------------------------------------------------------------
__global__ __launch_bounds__(256) void k_pack(const float* __restrict__ w1,
                                              const float* __restrict__ w2,
                                              unsigned short* __restrict__ whi,
                                              unsigned short* __restrict__ wlo) {
  unsigned t = blockIdx.x * 256u + threadIdx.x;   // 1,048,576 threads total
  unsigned lane = t & 31u; t >>= 5;
  unsigned kb   = t & 31u; t >>= 5;
  unsigned c    = t & 63u; t >>= 6;
  unsigned e    = t & 7u;  t >>= 3;
  unsigned mat  = t;                               // 0 = w1, 1 = w2
  const float* w = (mat ? w2 : w1) + (size_t)e * D_ * H_;
  int n = (int)(c * 16 + (lane & 15u));
  int khalf = (int)(lane >> 4) * 16;
  size_t obase = (((((size_t)mat * E_ + e) * 64 + c) * 32 + kb) * 32 + lane) * 16;
#pragma unroll
  for (int i = 0; i < 16; ++i) {
    int k = (int)kb * 32 + khalf + i;
    float v = w[(size_t)k * H_ + n];
    unsigned short hi = bf16_rn(v);
    unsigned short lo = bf16_rn(v - bf16_f32(hi));
    whi[obase + i] = hi;
    wlo[obase + i] = lo;
  }
}

// ---------------------------------------------------------------------------
// Gating: one wave per token (4 tokens per wave, 8 waves per block).
// Gate weights staged to LDS with gfx1250 async global->LDS copy.
// ---------------------------------------------------------------------------
__global__ __launch_bounds__(256) void k_gate(const float* __restrict__ x,
                                              const float* __restrict__ gw,
                                              int* cnt, float* zsum,
                                              float* dprox, float* dens1,
                                              int* tko, float* tkw) {
  __shared__ float s_gw[D_ * E_];                  // 32 KB
  int tid = threadIdx.x;
  // async copy: 2048 x 16B chunks, tracked by ASYNCcnt
  for (int i = tid; i < (D_ * E_) / 4; i += 256) {
    unsigned lds_off = (unsigned)(unsigned long long)(void*)(s_gw + i * 4);
    const float* src = gw + i * 4;
    asm volatile("global_load_async_to_lds_b128 %0, %1, off"
                 :: "v"(lds_off), "v"(src) : "memory");
  }
  asm volatile("s_wait_asynccnt 0x0" ::: "memory");
  __syncthreads();

  int wave = tid >> 5, lane = tid & 31;
  for (int s = 0; s < 4; ++s) {
    int token = ((int)blockIdx.x * 8 + wave) * 4 + s;
    const float* xr = x + (size_t)token * D_;
    float acc[E_];
#pragma unroll
    for (int e = 0; e < E_; ++e) acc[e] = 0.f;
    for (int d = lane; d < D_; d += 32) {
      float xv = xr[d];
#pragma unroll
      for (int e = 0; e < E_; ++e) acc[e] = fmaf(xv, s_gw[d * E_ + e], acc[e]);
    }
#pragma unroll
    for (int off = 16; off >= 1; off >>= 1)
#pragma unroll
      for (int e = 0; e < E_; ++e) acc[e] += __shfl_xor(acc[e], off, 32);
    if (lane == 0) {
      float mx = acc[0];
#pragma unroll
      for (int e = 1; e < E_; ++e) mx = fmaxf(mx, acc[e]);
      float p[E_], sum = 0.f;
#pragma unroll
      for (int e = 0; e < E_; ++e) { p[e] = expf(acc[e] - mx); sum += p[e]; }
      float z = logf(sum) + mx;
      atomicAdd(zsum, z * z);
      int b = token >> 11;                          // token / N_
      float inv = 1.f / sum;
      int i0 = 0;
#pragma unroll
      for (int e = 1; e < E_; ++e) if (p[e] > p[i0]) i0 = e;
      int i1 = (i0 == 0) ? 1 : 0;
#pragma unroll
      for (int e = 0; e < E_; ++e) if (e != i0 && p[e] > p[i1]) i1 = e;
#pragma unroll
      for (int e = 0; e < E_; ++e) atomicAdd(&dprox[b * E_ + e], p[e] * inv);
      atomicAdd(&dens1[b * E_ + i0], 1.0f);
      float p0 = p[i0] * inv, p1 = p[i1] * inv, s2 = p0 + p1;
      tko[token * 2 + 0] = i0; tko[token * 2 + 1] = i1;
      tkw[token * 2 + 0] = p0 / s2; tkw[token * 2 + 1] = p1 / s2;
      atomicAdd(&cnt[i0], 1); atomicAdd(&cnt[i1], 1);
    }
  }
}

// ---------------------------------------------------------------------------
// Scan: padded per-expert offsets + finalize scalar losses.
// ---------------------------------------------------------------------------
__global__ void k_scan(const int* cnt, int* offs, const float* dprox,
                       const float* dens1, const float* zsum,
                       float* out_tail) {
  int off = 0;
  for (int e = 0; e < E_; ++e) {
    offs[e] = off;
    off += ((cnt[e] + MT - 1) / MT) * MT;
  }
  offs[E_] = off;
  float bal = 0.f;
  for (int i = 0; i < B_ * E_; ++i)
    bal += (dprox[i] / (float)N_) * (dens1[i] / (float)N_);
  out_tail[0] = bal / (float)(B_ * E_) * (float)(E_ * E_);
  out_tail[1] = zsum[0] / (float)T_;
}

// ---------------------------------------------------------------------------
// Scatter tokens into expert-grouped list.
// ---------------------------------------------------------------------------
__global__ __launch_bounds__(256) void k_scatter(const int* tko, const float* tkw,
                                                 const int* offs, int* fill,
                                                 int* gtok, float* gcw) {
  int t = blockIdx.x * 256 + threadIdx.x;
  if (t >= T_) return;
#pragma unroll
  for (int s = 0; s < 2; ++s) {
    int e = tko[t * 2 + s];
    int pos = atomicAdd(&fill[e], 1);
    int g = offs[e] + pos;
    gtok[g] = t;
    gcw[g] = tkw[t * 2 + s];
  }
}

// A-layout packing helper: (row 0..31, K 0..1023) -> LDS u16 index.
static __device__ __forceinline__ int a_pack_idx(int row, int K) {
  int mtile = row >> 4;
  int kb    = K >> 5;
  int lane  = (row & 15) + (((K >> 4) & 1) << 4);
  int slot  = K & 15;
  return (((mtile * 32 + kb) * 32 + lane) * 16) + slot;
}

// ---------------------------------------------------------------------------
// Fused expert FFN: 32 tokens per block, 16 waves, bf16 hi/lo split WMMA.
// Each wave owns 4 column tiles and both M-tiles (B fragments reused 2x).
// ---------------------------------------------------------------------------
__global__ __launch_bounds__(512) void k_moe(const float* __restrict__ x,
                                             const unsigned short* __restrict__ whi,
                                             const unsigned short* __restrict__ wlo,
                                             const float* __restrict__ b1,
                                             const float* __restrict__ b2,
                                             const int* __restrict__ offs,
                                             const int* __restrict__ gtok,
                                             const float* __restrict__ gcw,
                                             float* __restrict__ out) {
  __shared__ unsigned short sAhi[2 * 32 * 32 * 16];   // 64 KB  X hi (A layout)
  __shared__ unsigned short sAlo[2 * 32 * 32 * 16];   // 64 KB  X lo
  __shared__ unsigned short sHhi[2 * 32 * 32 * 16];   // 64 KB  H hi
  __shared__ unsigned short sHlo[2 * 32 * 32 * 16];   // 64 KB  H lo
  __shared__ int   s_tok[MT];
  __shared__ float s_cw[MT];

  const int tile = blockIdx.x;
  const int ts = tile * MT;
  if (ts >= offs[E_]) return;                     // block-uniform exit
  int e = 0;
  while (e < E_ - 1 && ts >= offs[e + 1]) ++e;

  const int tid = threadIdx.x;
  if (tid < MT) { s_tok[tid] = gtok[ts + tid]; s_cw[tid] = gcw[ts + tid]; }
  __syncthreads();

  // stage gathered X rows -> packed A-layout bf16 hi/lo in LDS
  for (int it = 0; it < MT * D_ / 512; ++it) {
    int idx = it * 512 + tid;
    int row = idx >> 10, col = idx & (D_ - 1);
    int tok = s_tok[row];
    float v = (tok >= 0) ? x[(size_t)tok * D_ + col] : 0.f;
    unsigned short hi = bf16_rn(v);
    unsigned short lo = bf16_rn(v - bf16_f32(hi));
    int a = a_pack_idx(row, col);
    sAhi[a] = hi; sAlo[a] = lo;
  }
  __syncthreads();

  const int wave = tid >> 5, lane = tid & 31;    // 16 waves
  const int n_lo = lane & 15;
  const int mhalf = (lane >> 4) << 3;

  // ---------------- phase 1: H = gelu(X @ W1[e] + b1[e]) ----------------
  {
    v8f acc[2][4] = {};
    const size_t wbase = ((size_t)0 * E_ + e) * 64;
    for (int kb = 0; kb < 32; ++kb) {
      if (kb + 1 < 32) {
        size_t nfo = ((wbase + wave * 4) * 32 + (kb + 1)) * 512 + lane * 16;
        __builtin_prefetch(&whi[nfo], 0, 0);
        __builtin_prefetch(&wlo[nfo], 0, 0);
      }
      v16bf Ah[2], Al[2];
#pragma unroll
      for (int m = 0; m < 2; ++m) {
        Ah[m] = *(const v16bf*)&sAhi[((m * 32 + kb) * 32 + lane) * 16];
        Al[m] = *(const v16bf*)&sAlo[((m * 32 + kb) * 32 + lane) * 16];
      }
#pragma unroll
      for (int c = 0; c < 4; ++c) {
        int jt = wave * 4 + c;
        size_t fo = ((wbase + jt) * 32 + kb) * 512 + lane * 16;
        v16bf Bh = *(const v16bf*)&whi[fo];
        v16bf Bl = *(const v16bf*)&wlo[fo];
#pragma unroll
        for (int m = 0; m < 2; ++m) {
          acc[m][c] = __builtin_amdgcn_wmma_f32_16x16x32_bf16(false, Ah[m], false, Bh, (short)0, acc[m][c], false, false);
          acc[m][c] = __builtin_amdgcn_wmma_f32_16x16x32_bf16(false, Ah[m], false, Bl, (short)0, acc[m][c], false, false);
          acc[m][c] = __builtin_amdgcn_wmma_f32_16x16x32_bf16(false, Al[m], false, Bh, (short)0, acc[m][c], false, false);
        }
      }
    }
#pragma unroll
    for (int c = 0; c < 4; ++c) {
      int col = (wave * 4 + c) * 16 + n_lo;
      float bias = b1[e * H_ + col];
#pragma unroll
      for (int m = 0; m < 2; ++m) {
#pragma unroll
        for (int r = 0; r < 8; ++r) {
          int grow = m * 16 + r + mhalf;            // C layout: M = r + 8*(lane>=16)
          float hv = acc[m][c][r] + bias;
          float g = 0.5f * hv * (1.0f + erff(hv * 0.70710678118654752f));
          unsigned short hi = bf16_rn(g);
          unsigned short lo = bf16_rn(g - bf16_f32(hi));
          int a = a_pack_idx(grow, col);
          sHhi[a] = hi; sHlo[a] = lo;
        }
      }
    }
  }
  __syncthreads();

  // ---------------- phase 2: Y = H @ W2[e] + b2[e], combine -------------
  {
    v8f acc[2][4] = {};
    const size_t wbase = ((size_t)1 * E_ + e) * 64;
    for (int kb = 0; kb < 32; ++kb) {
      if (kb + 1 < 32) {
        size_t nfo = ((wbase + wave * 4) * 32 + (kb + 1)) * 512 + lane * 16;
        __builtin_prefetch(&whi[nfo], 0, 0);
        __builtin_prefetch(&wlo[nfo], 0, 0);
      }
      v16bf Ah[2], Al[2];
#pragma unroll
      for (int m = 0; m < 2; ++m) {
        Ah[m] = *(const v16bf*)&sHhi[((m * 32 + kb) * 32 + lane) * 16];
        Al[m] = *(const v16bf*)&sHlo[((m * 32 + kb) * 32 + lane) * 16];
      }
#pragma unroll
      for (int c = 0; c < 4; ++c) {
        int jt = wave * 4 + c;
        size_t fo = ((wbase + jt) * 32 + kb) * 512 + lane * 16;
        v16bf Bh = *(const v16bf*)&whi[fo];
        v16bf Bl = *(const v16bf*)&wlo[fo];
#pragma unroll
        for (int m = 0; m < 2; ++m) {
          acc[m][c] = __builtin_amdgcn_wmma_f32_16x16x32_bf16(false, Ah[m], false, Bh, (short)0, acc[m][c], false, false);
          acc[m][c] = __builtin_amdgcn_wmma_f32_16x16x32_bf16(false, Ah[m], false, Bl, (short)0, acc[m][c], false, false);
          acc[m][c] = __builtin_amdgcn_wmma_f32_16x16x32_bf16(false, Al[m], false, Bh, (short)0, acc[m][c], false, false);
        }
      }
    }
#pragma unroll
    for (int c = 0; c < 4; ++c) {
      int col = (wave * 4 + c) * 16 + n_lo;
      float bias = b2[e * H_ + col];
#pragma unroll
      for (int m = 0; m < 2; ++m) {
#pragma unroll
        for (int r = 0; r < 8; ++r) {
          int grow = m * 16 + r + mhalf;
          int tok = s_tok[grow];
          if (tok >= 0) {
            float val = s_cw[grow] * (acc[m][c][r] + bias);
            atomicAdd(&out[(size_t)tok * H_ + col], val);
          }
        }
      }
    }
  }
}

// ---------------------------------------------------------------------------
extern "C" void kernel_launch(void* const* d_in, const int* in_sizes, int n_in,
                              void* d_out, int out_size, void* d_ws, size_t ws_size,
                              hipStream_t stream) {
  const float* x  = (const float*)d_in[0];
  // d_in[1] = text, unused by the reference math
  const float* gw = (const float*)d_in[2];
  const float* w1 = (const float*)d_in[3];
  const float* b1 = (const float*)d_in[4];
  const float* w2 = (const float*)d_in[5];
  const float* b2 = (const float*)d_in[6];
  float* out = (float*)d_out;
  char* ws = (char*)d_ws;

  int*   cnt   = (int*)(ws + WS_CNT);
  int*   fill  = (int*)(ws + WS_FILL);
  int*   offs  = (int*)(ws + WS_OFF);
  float* zsum  = (float*)(ws + WS_ZSUM);
  float* dprox = (float*)(ws + WS_DPROX);
  float* dens1 = (float*)(ws + WS_DENS1);
  int*   tko   = (int*)(ws + WS_TOPKID);
  float* tkw   = (float*)(ws + WS_TOPKW);
  int*   gtok  = (int*)(ws + WS_GTOK);
  float* gcw   = (float*)(ws + WS_GCW);
  unsigned short* whi = (unsigned short*)(ws + WS_WHI);
  unsigned short* wlo = (unsigned short*)(ws + WS_WLO);

  // capture-safe resets
  hipMemsetAsync(ws, 0, 4096, stream);                              // counters/stats
  hipMemsetAsync(ws + WS_GTOK, 0xFF, (size_t)MAXE * 4, stream);     // tokens = -1
  hipMemsetAsync(d_out, 0, (size_t)(OUTN + 2) * sizeof(float), stream);

  k_pack<<<4096, 256, 0, stream>>>(w1, w2, whi, wlo);
  k_gate<<<T_ / 32, 256, 0, stream>>>(x, gw, cnt, zsum, dprox, dens1, tko, tkw);
  k_scan<<<1, 1, 0, stream>>>(cnt, offs, dprox, dens1, zsum, out + OUTN);
  k_scatter<<<T_ / 256, 256, 0, stream>>>(tko, tkw, offs, fill, gtok, gcw);
  k_moe<<<MAXTILES, 512, 0, stream>>>(x, whi, wlo, b1, b2, offs, gtok, gcw, out);
}
